// TimeIntervalAwareSelfAttention_60172491817561
// MI455X (gfx1250) — matrix operand order repaired and checked by reading, not compiled, not run on hardware
//
#include <hip/hip_runtime.h>
#include <hip/hip_bf16.h>

// Problem constants
#define B_    2
#define L_    512
#define H_    128
#define NH_   2
#define DH_   64
#define RV_   257          // REL_VOCAB
#define RPAD  272          // 257 padded to 17*16 for WMMA N tiling
#define WSTR  288          // wbar LDS stride / padded rel-vocab K extent (9*32)
#define SSTR  520          // scores LDS stride (512 + pad)

typedef __attribute__((ext_vector_type(16))) _Float16 v16h;
typedef __attribute__((ext_vector_type(8)))  float    v8f;

__device__ __forceinline__ v8f wmma16(v16h a, v16h b, v8f c) {
  // D = A(16x32 f16) * B(32x16 f16) + C(16x16 f32)
  return __builtin_amdgcn_wmma_f32_16x16x32_f16(false, a, false, b, (short)0, c,
                                                false, false);
}

// --- Fragment loaders per CDNA5 ISA 7.12.2 VGPR layouts (wave32) -----------
// A fragment 16x32 (MxK): all lanes carry M=lane&15;
//   lanes<16 -> K sub-base 0, lanes>=16 -> K sub-base 8 (within each 16-K group)

// A from row-major f32 (one-time consumers: raw inputs)
__device__ __forceinline__ v16h load_a_rm(const float* A, int row0, int lda,
                                          int k0, int lane) {
  const int m  = lane & 15;
  const int kb = (lane >> 4) << 3;                 // 0 or 8
  const float* p = A + (row0 + m) * lda + k0;
  v16h a;
#pragma unroll
  for (int e = 0; e < 16; ++e) {
    int kk = ((e >> 3) << 4) + kb + (e & 7);
    a[e] = (_Float16)p[kk];
  }
  return a;
}

// A from row-major f16 (two contiguous 16B runs per lane -> b128 loads)
__device__ __forceinline__ v16h load_a_rm_h(const _Float16* A, int row0,
                                            int lda, int k0, int lane) {
  const int m  = lane & 15;
  const int kb = (lane >> 4) << 3;
  const _Float16* p = A + (row0 + m) * lda + k0;
  v16h a;
#pragma unroll
  for (int e = 0; e < 16; ++e) a[e] = p[((e >> 3) << 4) + kb + (e & 7)];
  return a;
}

// B fragment 32x16 (KxN), B[k][n] = X[(n0+n)*ldx + k0+k] (transposed f16 src)
// -> two contiguous 16B runs per lane (table order).  rows (n) guarded by nlim.
__device__ __forceinline__ v16h load_b_tr_h(const _Float16* X, int n0, int ldx,
                                            int k0, int lane, int nlim) {
  const int n  = lane & 15;
  const int kb = (lane >> 4) << 3;                 // 0 or 8
  v16h b;
  if (n0 + n < nlim) {
    const _Float16* p = X + (n0 + n) * ldx + k0 + kb;
#pragma unroll
    for (int e = 0; e < 16; ++e) b[e] = p[((e >> 3) << 4) + (e & 7)];
  } else {
#pragma unroll
    for (int e = 0; e < 16; ++e) b[e] = (_Float16)0.f;
  }
  return b;
}

// ---------------------------------------------------------------------------
// Kernel 0: one-time f16 packing of weights / rel tables.
//   W16      : [4][128*128]  Wq,Wk,Wv,Wo
//   E_RK16   : [257*128]     row-major copy (B used K-contiguous along d)
//   E_RVT16  : [128][288]    transposed + zero-padded (B used K-contiguous
//                            along r; rows 257..287 are zero)
// ---------------------------------------------------------------------------
__global__ __launch_bounds__(256) void k_prep(
    const float* __restrict__ Wq, const float* __restrict__ Wk,
    const float* __restrict__ Wv, const float* __restrict__ Wo,
    const float* __restrict__ E_RK, const float* __restrict__ E_RV,
    _Float16* __restrict__ W16, _Float16* __restrict__ E_RK16,
    _Float16* __restrict__ E_RVT16) {
  int idx = blockIdx.x * 256 + threadIdx.x;
  const int NW = 4 * H_ * H_;                      // 65536
  const int NE = RV_ * H_;                         // 32896
  const int NT = H_ * WSTR;                        // 36864
  if (idx < NW) {
    int w = idx >> 14, j = idx & (H_ * H_ - 1);
    const float* src = (w == 0) ? Wq : (w == 1) ? Wk : (w == 2) ? Wv : Wo;
    W16[idx] = (_Float16)src[j];
  } else if (idx < NW + NE) {
    int j = idx - NW;
    E_RK16[j] = (_Float16)E_RK[j];
  } else if (idx < NW + NE + NT) {
    int j = idx - NW - NE;
    int n = j / WSTR, r = j % WSTR;                // n = feature, r = rel id
    E_RVT16[j] = (_Float16)((r < RV_) ? E_RV[r * H_ + n] : 0.f);
  }
}

// ---------------------------------------------------------------------------
// Kernel 1: fused QKV projections -> f16. 1536 wave-tiles.
//   Q16,K16 : [1024][128] row-major f16
//   VT16    : per batch [128][512] f16 (transposed: B frags of w@V contiguous)
// ---------------------------------------------------------------------------
__global__ __launch_bounds__(256) void k_qkv_proj(
    const float* __restrict__ query, const float* __restrict__ key,
    const float* __restrict__ value,
    const _Float16* __restrict__ W16,
    const float* __restrict__ bq, const float* __restrict__ bk,
    const float* __restrict__ bv,
    const float* __restrict__ E_PK, const float* __restrict__ E_PV,
    const int* __restrict__ poss,
    _Float16* __restrict__ Q16, _Float16* __restrict__ K16,
    _Float16* __restrict__ VT16) {
  const int lane = threadIdx.x & 31;
  const int wave = threadIdx.x >> 5;
  const int gw   = blockIdx.x * 8 + wave;          // 0..1535
  const int proj = gw >> 9;                        // 0:Q 1:K 2:V
  const int tile = gw & 511;
  const int m0 = (tile >> 3) * 16;
  const int n0 = (tile & 7) * 16;

  const float*    A    = (proj == 0) ? query : (proj == 1) ? key : value;
  const _Float16* W    = W16 + proj * H_ * H_;
  const float*    bias = (proj == 0) ? bq : (proj == 1) ? bk : bv;

  v8f c = {};
#pragma unroll
  for (int k0 = 0; k0 < H_; k0 += 32) {
    v16h a = load_a_rm(A, m0, H_, k0, lane);
    v16h b = load_b_tr_h(W, n0, H_, k0, lane, 1 << 30);
    c = wmma16(a, b, c);
  }
  const int n   = n0 + (lane & 15);
  const int mhi = (lane >> 4) << 3;
  const float bn = bias[n];
#pragma unroll
  for (int r = 0; r < 8; ++r) {
    int m = m0 + mhi + r;
    float v = c[r] + bn;
    if (proj == 0) {
      Q16[m * H_ + n] = (_Float16)v;
    } else if (proj == 1) {
      v += E_PK[poss[m] * H_ + n];
      K16[m * H_ + n] = (_Float16)v;
    } else {
      v += E_PV[poss[m] * H_ + n];
      // transposed store: VT16[b][n][l]
      VT16[(m >> 9) * (H_ * L_) + n * L_ + (m & (L_ - 1))] = (_Float16)v;
    }
  }
}

// ---------------------------------------------------------------------------
// Kernel 2: QR[b,h,l,r] = sum_d Qh[b,h,l,d] * E_RK[r, h*64+d]   (f32 result)
// per (b,h): [512 x 64] x [64 x 257(->272)] GEMM.  2176 wave-tiles.
// ---------------------------------------------------------------------------
__global__ __launch_bounds__(256) void k_qr(const _Float16* __restrict__ Q16,
                                            const _Float16* __restrict__ E_RK16,
                                            float* __restrict__ QR) {
  const int lane = threadIdx.x & 31;
  const int wave = threadIdx.x >> 5;
  const int gw   = blockIdx.x * 8 + wave;          // 0..2175
  const int bh   = gw / (32 * 17);                 // 0..3
  const int t    = gw % (32 * 17);
  const int m0   = (t / 17) * 16;                  // l tile
  const int n0   = (t % 17) * 16;                  // r tile (padded to 272)
  const int b = bh >> 1, h = bh & 1;

  v8f c = {};
#pragma unroll
  for (int k0 = 0; k0 < DH_; k0 += 32) {
    v16h a = load_a_rm_h(Q16, b * L_ + m0, H_, h * DH_ + k0, lane);
    v16h bf = load_b_tr_h(E_RK16, n0, H_, h * DH_ + k0, lane, RV_);
    c = wmma16(a, bf, c);
  }
  const int nn  = lane & 15;
  const int mhi = (lane >> 4) << 3;
  float* dst = QR + (bh * L_ + m0 + mhi) * RPAD + n0 + nn;
#pragma unroll
  for (int r = 0; r < 8; ++r) dst[r * RPAD] = c[r];
}

// ---------------------------------------------------------------------------
// Kernel 3: attention core. 1 block (256 thr / 8 waves) per (b, h, 16-row tile).
// ---------------------------------------------------------------------------
__global__ __launch_bounds__(256) void k_attn(
    const _Float16* __restrict__ Q16, const _Float16* __restrict__ K16,
    const _Float16* __restrict__ VT16, const float* __restrict__ QR,
    const _Float16* __restrict__ E_RVT16, const int* __restrict__ interval,
    _Float16* __restrict__ AO16) {
  __shared__ float S[16 * SSTR];                   // scores / weights tile (f32)
  __shared__ float WB[16 * WSTR];                  // per-row vocab buckets (f32)

  const int lane = threadIdx.x & 31;
  const int wave = threadIdx.x >> 5;
  const int tid  = threadIdx.x;
  const int b  = blockIdx.x >> 6;
  const int h  = (blockIdx.x >> 5) & 1;
  const int l0 = (blockIdx.x & 31) * 16;

  const int* itv = interval + (b * L_ + l0) * L_;  // [q][k]
  const float* QRb = QR + (((b * NH_) + h) * L_ + l0) * RPAD;

  // ---- step 1: S = Qh_tile @ Kh^T  (WMMA, f16 operands) ----
  v16h a0 = load_a_rm_h(Q16, b * L_ + l0, H_, h * DH_, lane);
  v16h a1 = load_a_rm_h(Q16, b * L_ + l0, H_, h * DH_ + 32, lane);
  for (int ct = wave; ct < 32; ct += 8) {          // 32 key tiles over 8 waves
    v8f c = {};
    v16h b0 = load_b_tr_h(K16, b * L_ + ct * 16, H_, h * DH_, lane, 1 << 30);
    c = wmma16(a0, b0, c);
    v16h b1 = load_b_tr_h(K16, b * L_ + ct * 16, H_, h * DH_ + 32, lane, 1 << 30);
    c = wmma16(a1, b1, c);
    const int nn = lane & 15, mhi = (lane >> 4) << 3;
#pragma unroll
    for (int r = 0; r < 8; ++r) S[(mhi + r) * SSTR + ct * 16 + nn] = c[r];
  }
  __syncthreads();

  // ---- step 2: add relative term via QR gather (L2-resident table) ----
  for (int idx = tid; idx < 16 * L_; idx += 256) {
    int q = idx >> 9, k = idx & (L_ - 1);
    S[q * SSTR + k] += QRb[q * RPAD + itv[q * L_ + k]];
  }
  __syncthreads();

  // ---- step 3: masked softmax (scale 1/sqrt(64)), wave per row ----
  for (int q = wave; q < 16; q += 8) {
    const int lq = l0 + q;
    float vals[16];
    float mx = -3.4e38f;
#pragma unroll
    for (int i = 0; i < 16; ++i) {
      int k = lane + i * 32;
      float s = (k <= lq) ? S[q * SSTR + k] * 0.125f : -3.4e38f;
      vals[i] = s;
      mx = fmaxf(mx, s);
    }
#pragma unroll
    for (int off = 16; off > 0; off >>= 1) mx = fmaxf(mx, __shfl_xor(mx, off, 32));
    float sum = 0.f;
#pragma unroll
    for (int i = 0; i < 16; ++i) {
      int k = lane + i * 32;
      float e = (k <= lq) ? __expf(vals[i] - mx) : 0.f;
      vals[i] = e;
      sum += e;
    }
#pragma unroll
    for (int off = 16; off > 0; off >>= 1) sum += __shfl_xor(sum, off, 32);
    const float inv = 1.0f / sum;
#pragma unroll
    for (int i = 0; i < 16; ++i) S[q * SSTR + lane + i * 32] = vals[i] * inv;
  }
  // zero the vocab buckets
  for (int idx = tid; idx < 16 * WSTR; idx += 256) WB[idx] = 0.f;
  __syncthreads();

  // ---- step 4/5: waves 0-3: out1 = w @ Vh (WMMA);  waves 4-7: scatter w ----
  v8f c1 = {};
  if (wave < 4) {
    const int n0 = wave * 16;                      // 4 tiles cover DH=64
    const _Float16* VTb = VT16 + b * (H_ * L_);
    for (int k0 = 0; k0 < L_; k0 += 32) {
      v16h a = load_a_rm(S, 0, SSTR, k0, lane);    // w tile from LDS (f32->f16)
      v16h bb = load_b_tr_h(VTb, h * DH_ + n0, L_, k0, lane, 1 << 30);
      c1 = wmma16(a, bb, c1);
    }
  } else {
#pragma unroll
    for (int i = 0; i < 4; ++i) {
      const int q = (wave - 4) * 4 + i;            // rows 0..15 across waves 4-7
      const int lq = l0 + q;
      for (int k = lane; k < L_; k += 32) {
        float w = S[q * SSTR + k];
        if (k <= lq && w != 0.f)
          atomicAdd(&WB[q * WSTR + itv[q * L_ + k]], w);   // ds_add_f32
      }
    }
  }
  __syncthreads();

  // ---- step 6/7: out2 = wbar @ E_RV_h (WMMA), fused store (f16) ----
  if (wave < 4) {
    const int n0 = wave * 16;
    v8f c2 = {};
    for (int k0 = 0; k0 < WSTR; k0 += 32) {
      v16h a = load_a_rm(WB, 0, WSTR, k0, lane);
      v16h bb = load_b_tr_h(E_RVT16, h * DH_ + n0, WSTR, k0, lane, 1 << 30);
      c2 = wmma16(a, bb, c2);
    }
    const int nn = lane & 15, mhi = (lane >> 4) << 3;
#pragma unroll
    for (int r = 0; r < 8; ++r)
      AO16[(b * L_ + l0 + mhi + r) * H_ + h * DH_ + n0 + nn] =
          (_Float16)(c1[r] + c2[r]);
  }
}

// ---------------------------------------------------------------------------
// Kernel 4: output projection + bias + NaN scrub. 512 wave-tiles.
// ---------------------------------------------------------------------------
__global__ __launch_bounds__(256) void k_oproj(const _Float16* __restrict__ AO16,
                                               const _Float16* __restrict__ Wo16,
                                               const float* __restrict__ bo,
                                               float* __restrict__ out) {
  const int lane = threadIdx.x & 31;
  const int wave = threadIdx.x >> 5;
  const int gw   = blockIdx.x * 8 + wave;          // 0..511
  const int m0 = (gw >> 3) * 16;
  const int n0 = (gw & 7) * 16;

  v8f c = {};
#pragma unroll
  for (int k0 = 0; k0 < H_; k0 += 32) {
    v16h a = load_a_rm_h(AO16, m0, H_, k0, lane);
    v16h b = load_b_tr_h(Wo16, n0, H_, k0, lane, 1 << 30);
    c = wmma16(a, b, c);
  }
  const int n   = n0 + (lane & 15);
  const int mhi = (lane >> 4) << 3;
  const float bn = bo[n];
#pragma unroll
  for (int r = 0; r < 8; ++r) {
    float v = c[r] + bn;
    v = (v != v) ? 0.f : v;                        // NaN -> 0
    out[(m0 + mhi + r) * H_ + n] = v;
  }
}

// ---------------------------------------------------------------------------
extern "C" void kernel_launch(void* const* d_in, const int* in_sizes, int n_in,
                              void* d_out, int out_size, void* d_ws,
                              size_t ws_size, hipStream_t stream) {
  (void)in_sizes; (void)n_in; (void)out_size; (void)ws_size;
  const float* query = (const float*)d_in[0];
  const float* key   = (const float*)d_in[1];
  const float* value = (const float*)d_in[2];
  const float* Wq = (const float*)d_in[3];
  const float* bq = (const float*)d_in[4];
  const float* Wk = (const float*)d_in[5];
  const float* bk = (const float*)d_in[6];
  const float* Wv = (const float*)d_in[7];
  const float* bv = (const float*)d_in[8];
  const float* Wo = (const float*)d_in[9];
  const float* bo = (const float*)d_in[10];
  const float* E_PK = (const float*)d_in[11];
  const float* E_PV = (const float*)d_in[12];
  const float* E_RK = (const float*)d_in[13];
  const float* E_RV = (const float*)d_in[14];
  const int* poss     = (const int*)d_in[15];
  const int* interval = (const int*)d_in[16];
  // d_in[17] (attn_mask) is deterministic causal tril — not needed.

  // ---- workspace layout (halves first, then f32 QR) ----
  _Float16* hws = (_Float16*)d_ws;
  const int NM = B_ * L_ * H_;                     // 131072
  _Float16* Q16     = hws;                         // 131072
  _Float16* K16     = Q16 + NM;                    // 131072
  _Float16* VT16    = K16 + NM;                    // 131072 ([b][128][512])
  _Float16* W16     = VT16 + NM;                   // 65536  (Wq,Wk,Wv,Wo)
  _Float16* E_RK16  = W16 + 4 * H_ * H_;           // 32896
  _Float16* E_RVT16 = E_RK16 + RV_ * H_;           // 36864  ([128][288])
  _Float16* AO16    = E_RVT16 + H_ * WSTR;         // 131072
  float*    QR      = (float*)(AO16 + NM);         // 557056 f32

  k_prep<<<529, 256, 0, stream>>>(Wq, Wk, Wv, Wo, E_RK, E_RV,
                                  W16, E_RK16, E_RVT16);
  k_qkv_proj<<<192, 256, 0, stream>>>(query, key, value, W16, bq, bk, bv,
                                      E_PK, E_PV, poss, Q16, K16, VT16);
  k_qr<<<272, 256, 0, stream>>>(Q16, E_RK16, QR);
  k_attn<<<128, 256, 0, stream>>>(Q16, K16, VT16, QR, E_RVT16, interval, AO16);
  k_oproj<<<64, 256, 0, stream>>>(AO16, W16 + 3 * H_ * H_, bo, (float*)d_out);
}
